// RoiTokenAttentionLayer_23691039604816
// MI455X (gfx1250) — compile-verified, matrix operand order checked by reading, not run on hardware
//
#include <hip/hip_runtime.h>
#include <cstdint>
#include <cstddef>

// ---------------------------------------------------------------------------
// Types for CDNA5 WMMA (wave32): v_wmma_f32_16x16x32_bf16
// ---------------------------------------------------------------------------
typedef __attribute__((ext_vector_type(16))) __bf16 v16bf;
typedef __attribute__((ext_vector_type(8)))  float  v8f;

union AFrag { v16bf v; unsigned u[8]; };
union BFrag { v16bf v; uint4 q[2]; };

__device__ __forceinline__ unsigned short f2bf(float f) {
  union { float f; unsigned u; } x; x.f = f;
  unsigned r = x.u + 0x7FFFu + ((x.u >> 16) & 1u);   // round-to-nearest-even
  return (unsigned short)(r >> 16);
}
__device__ __forceinline__ float bf2f(unsigned short h) {
  union { unsigned u; float f; } x; x.u = ((unsigned)h) << 16; return x.f;
}

// ---------------------------------------------------------------------------
// CDNA5 async copy: global -> LDS, 16B per lane, tracked by ASYNCcnt.
// Inline asm keeps this portable across ROCm7.2 / amdgpu-toolchain.
// ---------------------------------------------------------------------------
__device__ __forceinline__ void async_b128_to_lds(unsigned lds_off, const void* gaddr) {
  asm volatile("global_load_async_to_lds_b128 %0, %1, off"
               :: "v"(lds_off), "v"(gaddr) : "memory");
}
__device__ __forceinline__ void wait_async_0() {
  asm volatile("s_wait_asynccnt 0x0" ::: "memory");
}

// ---------------------------------------------------------------------------
// A fragment (16x32 bf16, row-major source): lane<16 -> row M=lane, K offsets
// {0..7,16..23}; lane>=16 -> row M=lane-16, K offsets {8..15,24..31}.
// Pairs are contiguous -> compiler merges into 2x global_load_b128.
// ---------------------------------------------------------------------------
__device__ __forceinline__ v16bf load_a_frag(const unsigned short* A, int lda,
                                             int m0, int k0, int lane) {
  const int r  = m0 + (lane & 15);
  const int hi = (lane >> 4) << 3;           // 0 or 8
  const unsigned* row = (const unsigned*)(A + (size_t)r * lda);
  AFrag f;
#pragma unroll
  for (int g = 0; g < 8; ++g) {
    int kg = k0 + hi + ((g < 4) ? (2 * g) : (16 + 2 * (g - 4)));
    f.u[g] = row[kg >> 1];
  }
  return f.v;
}

// ---------------------------------------------------------------------------
// Weight repack: fp32 (K x ld row-major, starting at column src_col) ->
// packed bf16 B-fragment layout; tile (n/16, k/32) stored as 32 lanes x 16
// contiguous bf16 (lane-major, 1KB per tile); K-tiles contiguous per n-tile.
// ---------------------------------------------------------------------------
__global__ void pack_w_kernel(const float* __restrict__ src, int ld, int src_col,
                              int K, int Ncols, int n_off,
                              unsigned short* __restrict__ dst) {
  int gid = blockIdx.x * 256 + threadIdx.x;
  if (gid >= K * Ncols) return;
  int nn = gid % Ncols;
  int k  = gid / Ncols;
  int n  = n_off + nn;
  float val = src[(size_t)k * ld + src_col + nn];
  int kw_  = k & 31;
  int half = (kw_ >> 3) & 1;                         // which 16-lane half
  int lane = half * 16 + (n & 15);
  int g    = ((kw_ >> 4) << 2) + ((kw_ & 7) >> 1);   // dword index 0..7
  int j    = 2 * g + (kw_ & 1);                      // bf16 index 0..15
  size_t tile = (size_t)(n >> 4) * (size_t)(K >> 5) + (size_t)(k >> 5);
  dst[tile * 512 + (size_t)lane * 16 + j] = f2bf(val);
}

// Fused bias vector for the big projection GEMM: [kb | vb | bk | bv]
__global__ void build_cbias_kernel(const float* __restrict__ mha_in_b,
                                   const float* __restrict__ bk,
                                   const float* __restrict__ bv,
                                   float* __restrict__ cb) {
  int n = blockIdx.x * 256 + threadIdx.x;
  if (n >= 3072) return;
  float v;
  if (n < 1536)      v = mha_in_b[768 + n];
  else if (n < 2304) v = bk[n - 1536];
  else               v = bv[n - 2304];
  cb[n] = v;
}

// ---------------------------------------------------------------------------
// Generic WMMA GEMM: C(MxN) = A(MxK, bf16) * B(packed bf16) + bias.
// Block: 256 threads (8 waves); block tile 128(M) x 64(N); wave tile 16x64.
// K-step = 64: an 8KB B slab (4 n-tiles x 2 k-tiles) is staged into LDS with
// async-to-LDS DMA, double-buffered. One barrier + one s_wait_asynccnt per
// step; the next stage is issued AFTER the barrier (all waves' LDS reads of
// the target buffer are already in VGPRs by then), and the last iteration is
// peeled so the inner loop is branch-free.
// MODE: 0 = f32 store, 1 = bf16 store, 2 = ReLU + bf16 store,
//       3 = f32 store of residual + acc.   Requires K % 64 == 0.
// ---------------------------------------------------------------------------
template <int MODE>
__global__ void gemm_bf16_kernel(const unsigned short* __restrict__ A, int lda,
                                 const unsigned short* __restrict__ Bp, int K,
                                 const float* __restrict__ bias,
                                 float* __restrict__ outF,
                                 unsigned short* __restrict__ outB,
                                 const float* __restrict__ residual, int ldc) {
  const int tid  = threadIdx.x;
  const int lane = tid & 31;
  const int wave = tid >> 5;
  const int m0 = blockIdx.x * 128 + wave * 16;
  const int n0 = blockIdx.y * 64;

  __shared__ __align__(16) unsigned short bsm[2][8 * 512];   // 2 x 8KB B slabs

  // Stage the 64(N) x 64(K) B slab for k0 into buffer `buf`: each n-chunk is
  // 2 consecutive 1KB packed tiles (2KB contiguous); 64 threads x 32B each.
  auto stage = [&](int buf, int k0) {
    int chunk = tid >> 6;                 // n-tile 0..3
    int sub   = tid & 63;                 // 32B unit within 2KB
    size_t tile = (size_t)((n0 >> 4) + chunk) * (size_t)(K >> 5) + (size_t)(k0 >> 5);
    const char* g = (const char*)Bp + tile * 1024 + (size_t)sub * 32;
    unsigned l = (unsigned)(size_t)(const void*)&bsm[buf][chunk * 1024 + sub * 16];
    async_b128_to_lds(l, g);
    async_b128_to_lds(l + 16, g + 16);
  };

  v8f acc[4] = {v8f{0}, v8f{0}, v8f{0}, v8f{0}};

  auto compute = [&](int buf, int k0) {
    __builtin_prefetch(A + (size_t)(m0 + (lane & 15)) * lda + k0 + 128, 0, 1);
#pragma unroll
    for (int kk = 0; kk < 2; ++kk) {
      v16bf a = load_a_frag(A, lda, m0, k0 + 32 * kk, lane);
#pragma unroll
      for (int t = 0; t < 4; ++t) {
        const uint4* p = (const uint4*)&bsm[buf][t * 1024 + kk * 512 + lane * 16];
        BFrag bf; bf.q[0] = p[0]; bf.q[1] = p[1];
        acc[t] = __builtin_amdgcn_wmma_f32_16x16x32_bf16(
            false, a, false, bf.v, (short)0, acc[t], false, false);
      }
    }
  };

  stage(0, 0);
  int buf = 0;
  int k0 = 0;
  for (; k0 < K - 64; k0 += 64) {        // branch-free steady state
    wait_async_0();                      // current buffer's DMA complete
    __syncthreads();                     // publish; prior reads already done
    stage(buf ^ 1, k0 + 64);             // overlap next DMA with this math
    compute(buf, k0);
    buf ^= 1;
  }
  wait_async_0();                        // peeled last step
  __syncthreads();
  compute(buf, k0);

  // D layout: VGPR i -> row m0 + i + (lane<16 ? 0 : 8); col = n0+16t+(lane&15)
  const int roff = (lane < 16) ? 0 : 8;
  const int col  = lane & 15;
#pragma unroll
  for (int t = 0; t < 4; ++t) {
    int n = n0 + 16 * t + col;
    float bias_v = bias[n];
#pragma unroll
    for (int i = 0; i < 8; ++i) {
      int row = m0 + i + roff;
      float v = acc[t][i] + bias_v;
      size_t off = (size_t)row * ldc + n;
      if (MODE == 0)      outF[off] = v;
      else if (MODE == 1) outB[off] = f2bf(v);
      else if (MODE == 2) outB[off] = f2bf(v > 0.f ? v : 0.f);
      else                outF[off] = residual[off] + v;
    }
  }
}

// ---------------------------------------------------------------------------
// LayerNorm over last dim (768), one row per block, output bf16.
// ---------------------------------------------------------------------------
__global__ void ln_kernel(const float* __restrict__ in,
                          const float* __restrict__ g,
                          const float* __restrict__ b,
                          unsigned short* __restrict__ outB) {
  const int row = blockIdx.x;
  const float* x = in + (size_t)row * 768;
  __shared__ float red[16];

  float v0 = x[threadIdx.x], v1 = x[threadIdx.x + 256], v2 = x[threadIdx.x + 512];
  float s  = v0 + v1 + v2;
  float sq = v0 * v0 + v1 * v1 + v2 * v2;
#pragma unroll
  for (int o = 16; o > 0; o >>= 1) {
    s  += __shfl_xor(s,  o, 32);
    sq += __shfl_xor(sq, o, 32);
  }
  int lane = threadIdx.x & 31, wid = threadIdx.x >> 5;
  if (lane == 0) { red[wid] = s; red[wid + 8] = sq; }
  __syncthreads();
  if (threadIdx.x == 0) {
    float ts = 0.f, tq = 0.f;
    for (int i = 0; i < 8; ++i) { ts += red[i]; tq += red[i + 8]; }
    red[0] = ts * (1.0f / 768.0f);
    red[1] = tq * (1.0f / 768.0f);
  }
  __syncthreads();
  float mean = red[0];
  float var  = red[1] - mean * mean;
  float rs   = rsqrtf(var + 1e-5f);
  unsigned short* o = outB + (size_t)row * 768;
  o[threadIdx.x]       = f2bf((v0 - mean) * rs * g[threadIdx.x]       + b[threadIdx.x]);
  o[threadIdx.x + 256] = f2bf((v1 - mean) * rs * g[threadIdx.x + 256] + b[threadIdx.x + 256]);
  o[threadIdx.x + 512] = f2bf((v2 - mean) * rs * g[threadIdx.x + 512] + b[threadIdx.x + 512]);
}

// ---------------------------------------------------------------------------
// Block-wide reduction helper (256 threads, 8 waves).  OP 0=sum, 1=max.
// ---------------------------------------------------------------------------
template <int OP>
__device__ __forceinline__ float block_reduce(float v, float* scratch) {
  int lane = threadIdx.x & 31, wid = threadIdx.x >> 5;
#pragma unroll
  for (int o = 16; o > 0; o >>= 1) {
    float w = __shfl_xor(v, o, 32);
    v = (OP == 0) ? (v + w) : fmaxf(v, w);
  }
  if (lane == 0) scratch[wid] = v;
  __syncthreads();
  float r = (OP == 0) ? 0.f : -3.4e38f;
  if (threadIdx.x < 8) r = scratch[threadIdx.x];
  if (wid == 0) {
#pragma unroll
    for (int o = 4; o > 0; o >>= 1) {
      float w = __shfl_xor(r, o, 32);
      r = (OP == 0) ? (r + w) : fmaxf(r, w);
    }
    if (lane == 0) scratch[0] = r;
  }
  __syncthreads();
  float res = scratch[0];
  __syncthreads();
  return res;
}

// ---------------------------------------------------------------------------
// MHA cross attention for one (n, h): scores(10x1024) in LDS, softmax,
// weighted sum over v.  k at proj col 0, v at proj col 768 (bf16, ld 3072).
// ---------------------------------------------------------------------------
__global__ void mha_attn_kernel(const float* __restrict__ q,      // (640,768) f32
                                const unsigned short* __restrict__ proj,
                                unsigned short* __restrict__ ca)  // (640,768) bf16
{
  const int n = blockIdx.x / 12;
  const int h = blockIdx.x % 12;
  __shared__ float qs[10][64];
  __shared__ float sc[10][1024];
  __shared__ float outsh[10][64];
  __shared__ float red[16];

  for (int i = threadIdx.x; i < 640; i += 256) {
    int t = i >> 6, d = i & 63;
    qs[t][d] = q[(size_t)(n * 10 + t) * 768 + h * 64 + d];
    outsh[t][d] = 0.f;
  }
  __syncthreads();

  // scores = q . k / sqrt(64)
  for (int p = threadIdx.x; p < 1024; p += 256) {
    const unsigned short* kr = proj + (size_t)(n * 1024 + p) * 3072 + h * 64;
    float acc[10];
#pragma unroll
    for (int t = 0; t < 10; ++t) acc[t] = 0.f;
    for (int d = 0; d < 64; ++d) {
      float kv = bf2f(kr[d]);
#pragma unroll
      for (int t = 0; t < 10; ++t) acc[t] += qs[t][d] * kv;
    }
#pragma unroll
    for (int t = 0; t < 10; ++t) sc[t][p] = acc[t] * 0.125f;
  }
  __syncthreads();

  // softmax over p for each t
  for (int t = 0; t < 10; ++t) {
    float m = -3.4e38f;
    for (int p = threadIdx.x; p < 1024; p += 256) m = fmaxf(m, sc[t][p]);
    m = block_reduce<1>(m, red);
    float s = 0.f;
    for (int p = threadIdx.x; p < 1024; p += 256) {
      float e = __expf(sc[t][p] - m);
      sc[t][p] = e; s += e;
    }
    s = block_reduce<0>(s, red);
    float inv = 1.0f / s;
    for (int p = threadIdx.x; p < 1024; p += 256) sc[t][p] *= inv;
  }
  __syncthreads();

  // ca = probs @ v  (10 x 64 per block)
  {
    int d  = threadIdx.x & 63;
    int pq = threadIdx.x >> 6;            // 0..3
    float acc[10];
#pragma unroll
    for (int t = 0; t < 10; ++t) acc[t] = 0.f;
    for (int p = pq; p < 1024; p += 4) {
      float vv = bf2f(proj[(size_t)(n * 1024 + p) * 3072 + 768 + h * 64 + d]);
#pragma unroll
      for (int t = 0; t < 10; ++t) acc[t] += sc[t][p] * vv;
    }
#pragma unroll
    for (int t = 0; t < 10; ++t) atomicAdd(&outsh[t][d], acc[t]);
  }
  __syncthreads();
  for (int i = threadIdx.x; i < 640; i += 256) {
    int t = i >> 6, d = i & 63;
    ca[(size_t)(n * 10 + t) * 768 + h * 64 + d] = f2bf(outsh[t][d]);
  }
}

// ---------------------------------------------------------------------------
// Assign attention with intra-temporal cumsum softmax, one batch n per block.
// kk at proj col 1536, vv at proj col 2304.  Writes probs (f32, d_out) and
// av = probs @ vv (bf16, for the wo GEMM).
// ---------------------------------------------------------------------------
__global__ void assign_attn_kernel(const float* __restrict__ qq,   // (640,768)
                                   const unsigned short* __restrict__ proj,
                                   float* __restrict__ probs_out,  // (640,1024)
                                   unsigned short* __restrict__ av)// (640,768)
{
  const int n = blockIdx.x;
  __shared__ float sc[10][1024];
  __shared__ float qs[768];
  __shared__ float red[16];
  const float scale = 0.03608439182435161f;   // 1/sqrt(768), TAU = 1

  // raw scores
  for (int t = 0; t < 10; ++t) {
    for (int i = threadIdx.x; i < 768; i += 256)
      qs[i] = qq[(size_t)(n * 10 + t) * 768 + i];
    __syncthreads();
    for (int p = threadIdx.x; p < 1024; p += 256) {
      const unsigned short* kr = proj + (size_t)(n * 1024 + p) * 3072 + 1536;
      float a = 0.f;
      for (int d = 0; d < 768; ++d) a += qs[d] * bf2f(kr[d]);
      sc[t][p] = a * scale;
    }
    __syncthreads();
  }

  // subtract cumulative sum of previous raw scores
  for (int p = threadIdx.x; p < 1024; p += 256) {
    float cs = 0.f;
#pragma unroll
    for (int t = 0; t < 10; ++t) {
      float raw = sc[t][p];
      sc[t][p] = raw - cs;
      cs += raw;
    }
  }
  __syncthreads();

  // softmax per t, write probs
  for (int t = 0; t < 10; ++t) {
    float m = -3.4e38f;
    for (int p = threadIdx.x; p < 1024; p += 256) m = fmaxf(m, sc[t][p]);
    m = block_reduce<1>(m, red);
    float s = 0.f;
    for (int p = threadIdx.x; p < 1024; p += 256) {
      float e = __expf(sc[t][p] - m);
      sc[t][p] = e; s += e;
    }
    s = block_reduce<0>(s, red);
    float inv = 1.0f / s;
    for (int p = threadIdx.x; p < 1024; p += 256) {
      float pr = sc[t][p] * inv;
      sc[t][p] = pr;
      probs_out[(size_t)(n * 10 + t) * 1024 + p] = pr;
    }
  }
  __syncthreads();

  // av = probs @ vv  (10 x 768), vv read once, 10 accumulators per thread
  for (int c = 0; c < 3; ++c) {
    int d = c * 256 + threadIdx.x;
    float acc[10];
#pragma unroll
    for (int t = 0; t < 10; ++t) acc[t] = 0.f;
    for (int p = 0; p < 1024; ++p) {
      float vv = bf2f(proj[(size_t)(n * 1024 + p) * 3072 + 2304 + d]);
#pragma unroll
      for (int t = 0; t < 10; ++t) acc[t] += sc[t][p] * vv;
    }
#pragma unroll
    for (int t = 0; t < 10; ++t)
      av[(size_t)(n * 10 + t) * 768 + d] = f2bf(acc[t]);
  }
}

// ---------------------------------------------------------------------------
// Host-side orchestration
// ---------------------------------------------------------------------------
extern "C" void kernel_launch(void* const* d_in, const int* in_sizes, int n_in,
                              void* d_out, int out_size, void* d_ws, size_t ws_size,
                              hipStream_t stream) {
  (void)in_sizes; (void)n_in; (void)out_size; (void)ws_size;

  const float* roi       = (const float*)d_in[0];
  const float* x         = (const float*)d_in[1];
  const float* ln_fm_g   = (const float*)d_in[2];
  const float* ln_fm_b   = (const float*)d_in[3];
  const float* ln_tok_g  = (const float*)d_in[4];
  const float* ln_tok_b  = (const float*)d_in[5];
  const float* ln_mca_g  = (const float*)d_in[6];
  const float* ln_mca_b  = (const float*)d_in[7];
  const float* ln_ca_g   = (const float*)d_in[8];
  const float* ln_ca_b   = (const float*)d_in[9];
  const float* ln_mo_g   = (const float*)d_in[10];
  const float* ln_mo_b   = (const float*)d_in[11];
  const float* mha_in_w  = (const float*)d_in[12];  // (768, 2304)
  const float* mha_in_b  = (const float*)d_in[13];
  const float* mha_out_w = (const float*)d_in[14];
  const float* mha_out_b = (const float*)d_in[15];
  const float* mca_w1    = (const float*)d_in[16];
  const float* mca_b1    = (const float*)d_in[17];
  const float* mca_w2    = (const float*)d_in[18];
  const float* mca_b2    = (const float*)d_in[19];
  const float* wq        = (const float*)d_in[20];
  const float* bq        = (const float*)d_in[21];
  const float* wk        = (const float*)d_in[22];
  const float* bk        = (const float*)d_in[23];
  const float* wv        = (const float*)d_in[24];
  const float* bv        = (const float*)d_in[25];
  const float* wo        = (const float*)d_in[26];
  const float* bo        = (const float*)d_in[27];
  const float* mo_w1     = (const float*)d_in[28];
  const float* mo_b1     = (const float*)d_in[29];
  const float* mo_w2     = (const float*)d_in[30];
  const float* mo_b2     = (const float*)d_in[31];

  float* outF = (float*)d_out;                     // (64,10,768) then probs
  float* probs_out = outF + (size_t)64 * 10 * 768;

  // --- workspace carve-up ---
  char* ws = (char*)d_ws;
  size_t off = 0;
  auto carve = [&](size_t bytes) {
    size_t r = off;
    off = (off + bytes + 255) & ~(size_t)255;
    return r;
  };
  unsigned short* xn    = (unsigned short*)(ws + carve((size_t)65536 * 768 * 2));
  unsigned short* Bbig  = (unsigned short*)(ws + carve((size_t)768 * 3072 * 2));
  unsigned short* proj  = (unsigned short*)(ws + carve((size_t)65536 * 3072 * 2));
  unsigned short* Bq    = (unsigned short*)(ws + carve((size_t)768 * 768 * 2));
  unsigned short* Bout  = (unsigned short*)(ws + carve((size_t)768 * 768 * 2));
  unsigned short* Bmca1 = (unsigned short*)(ws + carve((size_t)768 * 3072 * 2));
  unsigned short* Bmca2 = (unsigned short*)(ws + carve((size_t)3072 * 768 * 2));
  unsigned short* Bwq   = (unsigned short*)(ws + carve((size_t)768 * 768 * 2));
  unsigned short* Bwo   = (unsigned short*)(ws + carve((size_t)768 * 768 * 2));
  unsigned short* Bmo1  = (unsigned short*)(ws + carve((size_t)768 * 3072 * 2));
  unsigned short* Bmo2  = (unsigned short*)(ws + carve((size_t)3072 * 768 * 2));
  float*          cbias = (float*)(ws + carve((size_t)3072 * 4));
  unsigned short* lnb   = (unsigned short*)(ws + carve((size_t)640 * 768 * 2));
  unsigned short* cab   = (unsigned short*)(ws + carve((size_t)640 * 768 * 2));
  unsigned short* avb   = (unsigned short*)(ws + carve((size_t)640 * 768 * 2));
  unsigned short* t1b   = (unsigned short*)(ws + carve((size_t)640 * 3072 * 2));
  float*          qbuf  = (float*)(ws + carve((size_t)640 * 768 * 4));
  float*          R1    = (float*)(ws + carve((size_t)640 * 768 * 4));
  float*          R2    = (float*)(ws + carve((size_t)640 * 768 * 4));
  float*          R3    = (float*)(ws + carve((size_t)640 * 768 * 4));

  auto pgrid = [](int K, int N) { return dim3((unsigned)((K * N + 255) / 256)); };

  // 1) LN(x) -> bf16
  ln_kernel<<<65536, 256, 0, stream>>>(x, ln_fm_g, ln_fm_b, xn);

  // 2) Pack weights
  pack_w_kernel<<<pgrid(768, 1536), 256, 0, stream>>>(mha_in_w, 2304, 768, 768, 1536, 0,    Bbig); // kw|vw
  pack_w_kernel<<<pgrid(768, 768),  256, 0, stream>>>(wk,        768,   0, 768,  768, 1536, Bbig);
  pack_w_kernel<<<pgrid(768, 768),  256, 0, stream>>>(wv,        768,   0, 768,  768, 2304, Bbig);
  pack_w_kernel<<<pgrid(768, 768),  256, 0, stream>>>(mha_in_w, 2304,   0, 768,  768, 0, Bq);      // qw
  pack_w_kernel<<<pgrid(768, 768),  256, 0, stream>>>(mha_out_w, 768,   0, 768,  768, 0, Bout);
  pack_w_kernel<<<pgrid(768, 3072), 256, 0, stream>>>(mca_w1,   3072,   0, 768, 3072, 0, Bmca1);
  pack_w_kernel<<<pgrid(3072, 768), 256, 0, stream>>>(mca_w2,    768,   0, 3072, 768, 0, Bmca2);
  pack_w_kernel<<<pgrid(768, 768),  256, 0, stream>>>(wq,        768,   0, 768,  768, 0, Bwq);
  pack_w_kernel<<<pgrid(768, 768),  256, 0, stream>>>(wo,        768,   0, 768,  768, 0, Bwo);
  pack_w_kernel<<<pgrid(768, 3072), 256, 0, stream>>>(mo_w1,    3072,   0, 768, 3072, 0, Bmo1);
  pack_w_kernel<<<pgrid(3072, 768), 256, 0, stream>>>(mo_w2,     768,   0, 3072, 768, 0, Bmo2);
  build_cbias_kernel<<<12, 256, 0, stream>>>(mha_in_b, bk, bv, cbias);

  // 3) Big fused projection: proj = xn @ [kw|vw|wk|wv] + cbias  (bf16 out)
  gemm_bf16_kernel<1><<<dim3(512, 48), 256, 0, stream>>>(
      xn, 768, Bbig, 768, cbias, nullptr, proj, nullptr, 3072);

  // 4) ROI path
  ln_kernel<<<640, 256, 0, stream>>>(roi, ln_tok_g, ln_tok_b, lnb);
  gemm_bf16_kernel<0><<<dim3(5, 12), 256, 0, stream>>>(
      lnb, 768, Bq, 768, mha_in_b, qbuf, nullptr, nullptr, 768);          // q
  mha_attn_kernel<<<768, 256, 0, stream>>>(qbuf, proj, cab);              // ca (bf16)
  gemm_bf16_kernel<3><<<dim3(5, 12), 256, 0, stream>>>(
      cab, 768, Bout, 768, mha_out_b, R1, nullptr, roi, 768);             // R1 = roi + ca@Wo

  ln_kernel<<<640, 256, 0, stream>>>(R1, ln_mca_g, ln_mca_b, lnb);
  gemm_bf16_kernel<2><<<dim3(5, 48), 256, 0, stream>>>(
      lnb, 768, Bmca1, 768, mca_b1, nullptr, t1b, nullptr, 3072);         // relu(h@W1)
  gemm_bf16_kernel<3><<<dim3(5, 12), 256, 0, stream>>>(
      t1b, 3072, Bmca2, 3072, mca_b2, R2, nullptr, R1, 768);              // R2

  ln_kernel<<<640, 256, 0, stream>>>(R2, ln_ca_g, ln_ca_b, lnb);
  gemm_bf16_kernel<0><<<dim3(5, 12), 256, 0, stream>>>(
      lnb, 768, Bwq, 768, bq, qbuf, nullptr, nullptr, 768);               // qq
  assign_attn_kernel<<<64, 256, 0, stream>>>(qbuf, proj, probs_out, avb); // probs + av
  gemm_bf16_kernel<3><<<dim3(5, 12), 256, 0, stream>>>(
      avb, 768, Bwo, 768, bo, R3, nullptr, R2, 768);                      // R3

  ln_kernel<<<640, 256, 0, stream>>>(R3, ln_mo_g, ln_mo_b, lnb);
  gemm_bf16_kernel<2><<<dim3(5, 48), 256, 0, stream>>>(
      lnb, 768, Bmo1, 768, mo_b1, nullptr, t1b, nullptr, 3072);
  gemm_bf16_kernel<3><<<dim3(5, 12), 256, 0, stream>>>(
      t1b, 3072, Bmo2, 3072, mo_b2, outF, nullptr, R3, 768);              // final out
}
